// NeuralOmniRobot_29729763623474
// MI455X (gfx1250) — matrix operand order repaired and checked by reading, not compiled, not run on hardware
//
#include <hip/hip_runtime.h>
#include <hip/hip_bf16.h>
#include <math.h>

// ---------------------------------------------------------------------------
// NeuralOmniRobot rollout on MI455X (gfx1250, wave32, WMMA).
// One wave32 = 16 robots for all 512 steps. Transposed formulation:
//   H^T(64x16) = W(64xK) * X^T(Kx16)   => weights are static A-fragments in
// VGPRs, activations are B-fragments rebuilt each step with ds_swizzle
// half-wave swaps (no LDS round trip). Biases ride in as the WMMA C operand,
// broadcast-loaded from a tiny LDS table. atan2(sin t, cos t) is replaced by
// the mathematically identical wrap t - 2pi*rint(t/2pi)  (v_rndne_f32).
// ---------------------------------------------------------------------------

typedef __attribute__((ext_vector_type(16))) _Float16 v16h;
typedef __attribute__((ext_vector_type(8)))  float    v8f;

#define BROB   16384
#define TSTEPS 512
#define DTC    0.016f

union F16x16 { v16h h; unsigned int w[8]; };

__device__ __forceinline__ unsigned pack2h(float a, float b) {
  union { _Float16 h[2]; unsigned u; } p;
  p.h[0] = (_Float16)a; p.h[1] = (_Float16)b;
  return p.u;
}

// exchange a dword between lane L and lane L^16 (SWAPX16 group-of-32 swizzle)
__device__ __forceinline__ unsigned swap16(unsigned v) {
#if __has_builtin(__builtin_amdgcn_ds_swizzle)
  return (unsigned)__builtin_amdgcn_ds_swizzle((int)v, 0x401F);
#else
  return (unsigned)__shfl_xor((int)v, 16, 32);
#endif
}

__device__ __forceinline__ float fast_tanh(float x) {
#if __has_builtin(__builtin_amdgcn_tanhf)
  return __builtin_amdgcn_tanhf(x);          // gfx1250 v_tanh_f32
#else
  // tanh(x) = 1 - 2/(1+e^{2x}); hw exp2 + hw rcp, saturates correctly at +-1
  float e = __builtin_amdgcn_exp2f(x * 2.8853900817779268f); // 2*log2(e)
  return 1.0f - 2.0f * __builtin_amdgcn_rcpf(e + 1.0f);
#endif
}

__device__ __forceinline__ v8f wmma32f16(v16h a, v16h b, v8f c) {
  // D = A(16x32 f16) * B(32x16 f16) + C(16x16 f32)
  return __builtin_amdgcn_wmma_f32_16x16x32_f16(
      /*neg_a=*/false, a, /*neg_b=*/false, b,
      /*c_mod=*/(short)0, c, /*reuse_a=*/false, /*reuse_b=*/false);
}

// broadcast-load 8 consecutive f32 from LDS (uniform addr per half-wave)
__device__ __forceinline__ v8f ldsb(const float* p) {
  v8f c;
  #pragma unroll
  for (int i = 0; i < 8; ++i) c[i] = p[i];
  return c;
}

// Convert two f32 D-tiles (feature-in-VGPR, robot-in-lane) into one f16
// B-fragment for K-chunk c. B element e @ lane(l,g) needs feature
// f = 32c + 16g + e; source row lives in VGPR (f&7) of lane group (f>>3)&1
// of tile f>>4 = 2c+g  =>  pair-pack + half-wave swap + select.
__device__ __forceinline__ void buildB(const v8f* T, int c, bool hi, F16x16& B) {
  #pragma unroll
  for (int jj = 0; jj < 4; ++jj) {
    const int r = 2 * jj;
    unsigned qa  = pack2h(T[2*c  ][r], T[2*c  ][r+1]);
    unsigned qb  = pack2h(T[2*c+1][r], T[2*c+1][r+1]);
    unsigned sqa = swap16(qa);
    unsigned sqb = swap16(qb);
    B.w[jj]     = hi ? sqb : qa;   // K pair 32c+16g+2jj
    B.w[jj + 4] = hi ? qb  : sqa;  // K pair 32c+16g+8+2jj
  }
}

__global__ __launch_bounds__(256, 1)
void rollout_kernel(const float* __restrict__ st0, const float* __restrict__ cmd,
                    const float* __restrict__ W1,  const float* __restrict__ b1,
                    const float* __restrict__ W2,  const float* __restrict__ b2,
                    const float* __restrict__ W3,  const float* __restrict__ b3,
                    float* __restrict__ out) {
  __shared__ float sb[144];                       // b1[64] | b2[64] | b3 padded[16]
  const int tid = threadIdx.x;
  if (tid < 64)       sb[tid] = b1[tid];
  else if (tid < 128) sb[tid] = b2[tid - 64];
  else if (tid < 144) sb[tid] = (tid - 128 < 3) ? b3[tid - 128] : 0.0f;
  __syncthreads();

  const int lane = tid & 31;
  const int wave = tid >> 5;
  const int l    = lane & 15;                     // robot slot / matrix column
  const int g    = lane >> 4;                     // half-wave (K group)
  const bool lo  = (g == 0);
  const int  r   = (blockIdx.x * 8 + wave) * 16 + l;   // robot id

  // ---- static f16 A-fragments (resident in VGPRs for the whole rollout) ----
  // A element e @ lane(l,g): row M = l, col K = e + 16g
  F16x16 W1a[4], W2a[4][2], W3a[2];
  #pragma unroll
  for (int t4 = 0; t4 < 4; ++t4) {
    #pragma unroll
    for (int e = 0; e < 16; ++e) {
      const int kk = e + 16 * g;
      W1a[t4].h[e]    = (kk < 6) ? (_Float16)W1[(16*t4 + l)*6 + kk] : (_Float16)0.0f;
      W2a[t4][0].h[e] = (_Float16)W2[(16*t4 + l)*64 + kk];
      W2a[t4][1].h[e] = (_Float16)W2[(16*t4 + l)*64 + 32 + kk];
    }
  }
  #pragma unroll
  for (int cc = 0; cc < 2; ++cc) {
    #pragma unroll
    for (int e = 0; e < 16; ++e) {
      const int kk = e + 16 * g;
      W3a[cc].h[e] = (l < 3) ? (_Float16)W3[l*64 + 32*cc + kk] : (_Float16)0.0f;
    }
  }

  // ---- per-robot state in registers (valid in lanes 0-15; zero-init so the
  //      mirror lanes compute finite garbage that never reaches a WMMA) ----
  float x = 0.f, y = 0.f, th = 0.f, vxw = 0.f, vyw = 0.f, om = 0.f;
  if (lo) {
    const float* sp = st0 + (size_t)r * 6;
    x = sp[0]; y = sp[1]; th = sp[2]; vxw = sp[3]; vyw = sp[4]; om = sp[5];
    float* op = out + (size_t)r * (TSTEPS + 1) * 6;   // states[:,0] = initial
    op[0] = x; op[1] = y; op[2] = th; op[3] = vxw; op[4] = vyw; op[5] = om;
  }
  const float* cbase = cmd + (size_t)r * TSTEPS * 3;
  float*       obase = out + (size_t)r * (TSTEPS + 1) * 6;

  #pragma unroll 1
  for (int t = 0; t < TSTEPS; ++t) {
    float c0 = 0.f, c1 = 0.f, c2 = 0.f;
    if (lo) {
      const float* cp = cbase + t * 3;
      c0 = cp[0]; c1 = cp[1]; c2 = cp[2];
    }
    if (t + 16 < TSTEPS) __builtin_prefetch(cbase + (t + 16) * 3, 0, 1);

    // world -> body velocity
    const float cth = __cosf(th), sth = __sinf(th);
    const float vxb =  vxw * cth + vyw * sth;
    const float vyb = -vxw * sth + vyw * cth;

    // B1 = nn_in^T (K=6 of 32); only half-wave 0 carries data
    F16x16 B1;
    #pragma unroll
    for (int i = 0; i < 8; ++i) B1.w[i] = 0u;
    B1.w[0] = lo ? pack2h(vxb, vyb) : 0u;
    B1.w[1] = lo ? pack2h(om,  c0 ) : 0u;
    B1.w[2] = lo ? pack2h(c1,  c2 ) : 0u;

    // layer 1: H1^T = tanh(W1 * nn_in^T + b1)
    v8f H1[4];
    #pragma unroll
    for (int tt = 0; tt < 4; ++tt) {
      v8f cacc = ldsb(&sb[16 * tt + 8 * g]);          // bias as C operand
      H1[tt] = wmma32f16(W1a[tt].h, B1.h, cacc);
      #pragma unroll
      for (int rr = 0; rr < 8; ++rr) H1[tt][rr] = fast_tanh(H1[tt][rr]);
    }

    // layer 2: H2^T = tanh(W2 * H1^T + b2)
    F16x16 B2[2];
    buildB(H1, 0, !lo, B2[0]);
    buildB(H1, 1, !lo, B2[1]);
    v8f H2[4];
    #pragma unroll
    for (int tt = 0; tt < 4; ++tt) {
      v8f cacc = ldsb(&sb[64 + 16 * tt + 8 * g]);
      cacc   = wmma32f16(W2a[tt][0].h, B2[0].h, cacc);
      H2[tt] = wmma32f16(W2a[tt][1].h, B2[1].h, cacc);
      #pragma unroll
      for (int rr = 0; rr < 8; ++rr) H2[tt][rr] = fast_tanh(H2[tt][rr]);
    }

    // layer 3: accel^T = W3 * H2^T + b3  (rows 0..2 valid, in half-wave 0)
    F16x16 B3[2];
    buildB(H2, 0, !lo, B3[0]);
    buildB(H2, 1, !lo, B3[1]);
    v8f a3 = ldsb(&sb[128 + 8 * g]);
    a3 = wmma32f16(W3a[0].h, B3[0].h, a3);
    a3 = wmma32f16(W3a[1].h, B3[1].h, a3);

    // integrate (faithful to the reference, including its nvyw quirk).
    // atan2(sin t, cos t) == wrap to [-pi,pi]: t - 2pi*rint(t/2pi).
    const float nvxb = vxb + a3[0] * DTC;
    const float nvyb = vyb + a3[1] * DTC;
    const float nom  = om  + a3[2] * DTC;
    float nth = th + nom * DTC;
    nth = nth - 6.283185307179586f * __builtin_rintf(nth * 0.15915494309189535f);
    const float cn = __cosf(nth), sn = __sinf(nth);
    const float nvxw = nvxb * cn - nvyb * sn;
    const float nvyw = nvxb * sn + nvxb * cn;   // reference uses nvxb twice
    x += nvxw * DTC; y += nvyw * DTC;
    th = nth; vxw = nvxw; vyw = nvyw; om = nom;

    if (lo) {
      float* op = obase + (size_t)(t + 1) * 6;   // 24B-aligned -> 8B stores ok
      *(float2*)(op + 0) = make_float2(x,   y);
      *(float2*)(op + 2) = make_float2(th,  vxw);
      *(float2*)(op + 4) = make_float2(vyw, om);
    }
  }
}

extern "C" void kernel_launch(void* const* d_in, const int* in_sizes, int n_in,
                              void* d_out, int out_size, void* d_ws, size_t ws_size,
                              hipStream_t stream) {
  (void)in_sizes; (void)n_in; (void)out_size; (void)d_ws; (void)ws_size;
  const float* st0 = (const float*)d_in[0];
  const float* cmd = (const float*)d_in[1];
  const float* W1  = (const float*)d_in[2];
  const float* b1  = (const float*)d_in[3];
  const float* W2  = (const float*)d_in[4];
  const float* b2  = (const float*)d_in[5];
  const float* W3  = (const float*)d_in[6];
  const float* b3  = (const float*)d_in[7];
  float* out = (float*)d_out;

  dim3 block(256);                 // 8 waves -> 128 robots per block
  dim3 grid(BROB / 128);           // 128 blocks, 1024 waves total
  rollout_kernel<<<grid, block, 0, stream>>>(st0, cmd, W1, b1, W2, b2, W3, b3, out);
}
